// LinearAttention_50989851738203
// MI455X (gfx1250) — matrix operand order repaired and checked by reading, not compiled, not run on hardware
//
#include <hip/hip_runtime.h>

typedef __attribute__((ext_vector_type(16))) _Float16 v16h;
typedef __attribute__((ext_vector_type(8)))  _Float16 v8h;
typedef __attribute__((ext_vector_type(8)))  float    v8f;

#define NB 16          // batch
#define C 256          // channels
#define NPIX 4096      // H*W
#define HID 256        // heads*dim_head
#define O3 768         // 3*HID
#define NHEADS 8
#define DHEAD 32
#define SCALE 0.17677669529663687f
#define EPS 1e-12f

// ---------------- WMMA fragment helpers (f16 16x16x32) ----------------
// A (16xK=32, row-major source, ld = row stride in halves):
//   lanes 0-15: row M=lane, K = {k0..k0+7} U {k0+16..k0+23}
//   lanes16-31: row M=lane-16, K = {k0+8..k0+15} U {k0+24..k0+31}
static __device__ inline v16h load_a_frag(const _Float16* base, int ld, int row0,
                                          int k0, int lane) {
    int m  = lane & 15;
    int hh = lane >> 4;
    const _Float16* p = base + (size_t)(row0 + m) * ld + k0 + hh * 8;
    v8h lo = *(const v8h*)p;
    v8h hi = *(const v8h*)(p + 16);
    return __builtin_shufflevector(lo, hi, 0,1,2,3,4,5,6,7,8,9,10,11,12,13,14,15);
}
// B (K=32 x 16, source is [col][k] with ld = col stride in halves):
//   lanes 0-15: col N=lane, K = {k0..k0+7} U {k0+16..k0+23} ; lanes16-31: col N=lane-16, K = {k0+8..k0+15} U {k0+24..k0+31}
static __device__ inline v16h load_b_frag(const _Float16* base, int ld, int col0,
                                          int k0, int lane) {
    int n  = lane & 15;
    int hh = lane >> 4;
    const _Float16* p = base + (size_t)(col0 + n) * ld + k0 + hh * 8;
    v8h lo = *(const v8h*)p;
    v8h hi = *(const v8h*)(p + 16);
    return __builtin_shufflevector(lo, hi, 0,1,2,3,4,5,6,7,8,9,10,11,12,13,14,15);
}
static __device__ inline v8f wmma_f16(v16h a, v16h b, v8f c) {
    return __builtin_amdgcn_wmma_f32_16x16x32_f16(false, a, false, b, (short)0, c,
                                                  false, false);
}

// ---------------- Kernel 0: weight f32 -> f16 ----------------
__global__ __launch_bounds__(256) void cvt_weights(const float* __restrict__ wq,
                                                   const float* __restrict__ wo,
                                                   _Float16* __restrict__ wq16,
                                                   _Float16* __restrict__ wo16) {
    int i = blockIdx.x * 256 + threadIdx.x;
    if (i < O3 * C)  wq16[i] = (_Float16)wq[i];
    if (i < HID * C) wo16[i] = (_Float16)wo[i];
}

// ---- Kernel 1: RMSNorm + QKV projection + fused q-softmax over dim_head ----
// Per block: 64 pixels. GEMM D[768 x 64] = Wqkv * xn^T. q tiles (rows 0..255)
// stay in registers, get spilled to LDS after the GEMM, softmaxed over d=32
// per (pixel, head) and stored; k/v stream straight to the workspace.
__global__ __launch_bounds__(256) void qkv_kernel(const float* __restrict__ x,
                                                  const float* __restrict__ g1,
                                                  const _Float16* __restrict__ wq16,
                                                  float* __restrict__ qkv) {
    __shared__ __align__(32) char smem[65536];
    _Float16* xs  = (_Float16*)smem;           // [pixel][channel] f16, 32 KB (phase 1)
    float* red    = (float*)(smem + 32768);    // 256 floats        (phase 1)
    float* invn   = (float*)(smem + 33792);    // 64 floats         (phase 1)
    float* qls    = (float*)smem;              // [channel][pixel] f32, 64 KB (phase 2)
    const int tid = threadIdx.x, lane = tid & 31, wave = tid >> 5;
    const int b = blockIdx.y;
    const int pix0 = blockIdx.x * 64;
    const float* xb = x + (size_t)b * C * NPIX + pix0;

    // --- per-pixel sum of squares over C (4 threads per pixel) ---
    int p = tid & 63, part = tid >> 6;
    float s = 0.f;
    for (int j = 0; j < 64; ++j) {
        float v = xb[(size_t)(part * 64 + j) * NPIX + p];
        s += v * v;
    }
    red[p * 4 + part] = s;
    __syncthreads();
    if (tid < 64) {
        float t = red[tid * 4] + red[tid * 4 + 1] + red[tid * 4 + 2] + red[tid * 4 + 3];
        invn[tid] = 16.0f / fmaxf(sqrtf(t), EPS);   // sqrt(C)=16
    }
    __syncthreads();
    for (int j = 0; j < 64; ++j) {
        int c = part * 64 + j;
        float v = xb[(size_t)c * NPIX + p];
        xs[p * C + c] = (_Float16)(v * invn[p] * g1[c]);
    }
    __syncthreads();

    // --- GEMM: each wave owns M-strips mt = wave + 8*i; A frag feeds 4 WMMAs ---
    float* outb = qkv + (size_t)b * O3 * NPIX;
    const int hh = lane >> 4, ln = lane & 15;
    const v8f z = {0.f, 0.f, 0.f, 0.f, 0.f, 0.f, 0.f, 0.f};
    v8f qacc[2][4];
#pragma unroll
    for (int i = 0; i < 6; ++i) {
        int mt = wave + 8 * i;
        v8f acc[4] = {z, z, z, z};
#pragma unroll
        for (int kb = 0; kb < 8; ++kb) {
            v16h a = load_a_frag(wq16, C, mt * 16, kb * 32, lane);
#pragma unroll
            for (int nt = 0; nt < 4; ++nt) {
                v16h bb = load_b_frag(xs, C, nt * 16, kb * 32, lane);
                acc[nt] = wmma_f16(a, bb, acc[nt]);
            }
        }
        if (i < 2) {            // q rows (0..255): keep for fused softmax
#pragma unroll
            for (int nt = 0; nt < 4; ++nt) qacc[i][nt] = acc[nt];
        } else {                // k/v rows (256..767): stream to workspace
#pragma unroll
            for (int nt = 0; nt < 4; ++nt)
#pragma unroll
                for (int r = 0; r < 8; ++r) {
                    int row = mt * 16 + hh * 8 + r;
                    outb[(size_t)row * NPIX + pix0 + nt * 16 + ln] = acc[nt][r];
                }
        }
    }
    __syncthreads();            // xs dead -> qls may overwrite

    // --- spill q tiles to LDS as [channel][pixel] ---
#pragma unroll
    for (int i = 0; i < 2; ++i)
#pragma unroll
        for (int nt = 0; nt < 4; ++nt)
#pragma unroll
            for (int r = 0; r < 8; ++r) {
                int row = (wave + 8 * i) * 16 + hh * 8 + r;
                qls[row * 64 + nt * 16 + ln] = qacc[i][nt][r];
            }
    __syncthreads();

    // --- softmax over d=32 per (pixel, head); lane->pixel => bank-conflict-free ---
    int sp = tid & 63, h0 = tid >> 6;
    float* qout = outb + pix0 + sp;
#pragma unroll
    for (int hi = 0; hi < 2; ++hi) {
        int h = h0 + hi * 4;
        float v[DHEAD];
        float m = -1e30f;
#pragma unroll
        for (int d = 0; d < DHEAD; ++d) {
            v[d] = qls[(h * DHEAD + d) * 64 + sp];
            m = fmaxf(m, v[d]);
        }
        float ssum = 0.f;
#pragma unroll
        for (int d = 0; d < DHEAD; ++d) { v[d] = __expf(v[d] - m); ssum += v[d]; }
        float inv = SCALE / ssum;
#pragma unroll
        for (int d = 0; d < DHEAD; ++d)
            qout[(size_t)(h * DHEAD + d) * NPIX] = v[d] * inv;
    }
}

// ---------------- Kernel 2: k softmax stats + context = softmax(k) @ v^T ----
__global__ __launch_bounds__(256) void context_kernel(const float* __restrict__ qkv,
                                                      _Float16* __restrict__ ctxT) {
    __shared__ float ctxp[8 * 1024];   // per-wave 32x32 partials, 32 KB
    __shared__ float red[256];
    __shared__ float mrow[DHEAD];
    __shared__ float srow[DHEAD];
    const int tid = threadIdx.x, lane = tid & 31, wave = tid >> 5;
    const int bh = blockIdx.x, b = bh >> 3, h = bh & 7;
    const float* kbase = qkv + ((size_t)b * O3 + HID + h * DHEAD) * NPIX;
    const float* vbase = qkv + ((size_t)b * O3 + 2 * HID + h * DHEAD) * NPIX;

    // row max over n (8 threads per row of 4096), with stream prefetch
    int d = tid >> 3, part = tid & 7;
    const float* kr = kbase + (size_t)d * NPIX + part * 512;
    float m = -1e30f;
    for (int j0 = 0; j0 < 512; j0 += 64) {
        __builtin_prefetch(kr + j0 + 256, 0, 1);
#pragma unroll
        for (int j = 0; j < 64; ++j) m = fmaxf(m, kr[j0 + j]);
    }
    red[tid] = m;
    __syncthreads();
    if (tid < DHEAD) {
        float mm = red[tid * 8];
        for (int i = 1; i < 8; ++i) mm = fmaxf(mm, red[tid * 8 + i]);
        mrow[tid] = mm;
    }
    __syncthreads();
    // row exp-sum
    float md = mrow[d];
    float ss = 0.f;
    for (int j0 = 0; j0 < 512; j0 += 64) {
        __builtin_prefetch(kr + j0 + 256, 0, 1);
#pragma unroll
        for (int j = 0; j < 64; ++j) ss += __expf(kr[j0 + j] - md);
    }
    red[tid] = ss;
    __syncthreads();
    if (tid < DHEAD) {
        float t = 0.f;
        for (int i = 0; i < 8; ++i) t += red[tid * 8 + i];
        srow[tid] = t;
    }
    __syncthreads();

    // context[32x32] = exp(k - m) @ v^T over K=4096, split over 8 waves
    const int hh = lane >> 4, ln = lane & 15;
    v8f z = {0.f, 0.f, 0.f, 0.f, 0.f, 0.f, 0.f, 0.f};
    v8f acc[2][2] = {{z, z}, {z, z}};
    for (int kb = wave; kb < 128; kb += 8) {
        v16h afr[2], bfr[2];
#pragma unroll
        for (int mt = 0; mt < 2; ++mt) {
            int row = mt * 16 + ln;
            const float* kp = kbase + (size_t)row * NPIX + kb * 32 + hh * 8;
            float mv = mrow[row];
            v16h a;
#pragma unroll
            for (int i = 0; i < 8; ++i) a[i]     = (_Float16)__expf(kp[i] - mv);
#pragma unroll
            for (int i = 0; i < 8; ++i) a[8 + i] = (_Float16)__expf(kp[16 + i] - mv);
            afr[mt] = a;
        }
#pragma unroll
        for (int nt = 0; nt < 2; ++nt) {
            int col = nt * 16 + ln;
            const float* vp = vbase + (size_t)col * NPIX + kb * 32 + hh * 8;
            v16h bb;
#pragma unroll
            for (int i = 0; i < 8; ++i) { bb[i] = (_Float16)vp[i]; bb[8 + i] = (_Float16)vp[16 + i]; }
            bfr[nt] = bb;
        }
#pragma unroll
        for (int mt = 0; mt < 2; ++mt)
#pragma unroll
            for (int nt = 0; nt < 2; ++nt)
                acc[mt][nt] = wmma_f16(afr[mt], bfr[nt], acc[mt][nt]);
    }
    // cross-wave reduce, normalize by row sums, store transposed f16
#pragma unroll
    for (int mt = 0; mt < 2; ++mt)
#pragma unroll
        for (int nt = 0; nt < 2; ++nt)
#pragma unroll
            for (int r = 0; r < 8; ++r)
                ctxp[wave * 1024 + (mt * 16 + hh * 8 + r) * 32 + nt * 16 + ln] =
                    acc[mt][nt][r];
    __syncthreads();
    for (int idx = tid; idx < 1024; idx += 256) {
        int dd = idx >> 5, e = idx & 31;
        float t = 0.f;
        for (int w = 0; w < 8; ++w) t += ctxp[w * 1024 + idx];
        ctxT[(size_t)bh * 1024 + e * 32 + dd] = (_Float16)(t / srow[dd]);
    }
}

// ---------------- Kernel 3: attention out + out-proj + RMSNorm + residual ---
__global__ __launch_bounds__(256) void out_kernel(const float* __restrict__ qkv,
                                                  const _Float16* __restrict__ ctxT,
                                                  const _Float16* __restrict__ wo16,
                                                  const float* __restrict__ b_out,
                                                  const float* __restrict__ g2,
                                                  const float* __restrict__ x,
                                                  float* __restrict__ out) {
    __shared__ __align__(32) char smem[34048];
    _Float16* qs  = (_Float16*)smem;             // [32 px][256 ch] 16 KB (phase 1)
    _Float16* att = (_Float16*)(smem + 16384);   // [32 px][256 ch] 16 KB (phase 2)
    float*    ys  = (float*)smem;                // [32 px][256 ch] 32 KB (phase 3, overlaps)
    float*    red = (float*)(smem + 32768);      // 288 floats
    const int tid = threadIdx.x, lane = tid & 31, wave = tid >> 5;
    const int b = blockIdx.y;
    const int pix0 = blockIdx.x * 32;
    const int hh = lane >> 4, ln = lane & 15;
    const v8f z = {0.f, 0.f, 0.f, 0.f, 0.f, 0.f, 0.f, 0.f};

    // stage q tile transposed into LDS (f16)
    const float* qbase = qkv + (size_t)b * O3 * NPIX + pix0;
    for (int idx = tid; idx < 32 * C; idx += 256) {
        int p = idx & 31, c = idx >> 5;
        qs[p * C + c] = (_Float16)qbase[(size_t)c * NPIX + p];
    }
    __syncthreads();

    // attention out: att[e_global, p] = sum_d ctxT[e,d] * q[d,p]; A reused over nt
#pragma unroll
    for (int i = 0; i < 2; ++i) {
        int mt = wave + 8 * i;
        int head = mt >> 1;
        const _Float16* actx = ctxT + ((size_t)b * NHEADS + head) * 1024;
        v16h a = load_a_frag(actx, 32, (mt & 1) * 16, 0, lane);
#pragma unroll
        for (int nt = 0; nt < 2; ++nt) {
            v16h bb = load_b_frag(qs + head * DHEAD, C, nt * 16, 0, lane);
            v8f acc = wmma_f16(a, bb, z);
#pragma unroll
            for (int r = 0; r < 8; ++r)
                att[(nt * 16 + ln) * C + mt * 16 + hh * 8 + r] = (_Float16)acc[r];
        }
    }
    __syncthreads();

    // out projection: y[256 x 32] = Wout[256 x 256] * att; A frag feeds 2 WMMAs
    v8f accs[2][2];
#pragma unroll
    for (int i = 0; i < 2; ++i) {
        int mt = wave + 8 * i;
        v8f a0 = z, a1 = z;
#pragma unroll
        for (int kb = 0; kb < 8; ++kb) {
            v16h a  = load_a_frag(wo16, C, mt * 16, kb * 32, lane);
            v16h b0 = load_b_frag(att, C, 0,  kb * 32, lane);
            v16h b1 = load_b_frag(att, C, 16, kb * 32, lane);
            a0 = wmma_f16(a, b0, a0);
            a1 = wmma_f16(a, b1, a1);
        }
        accs[i][0] = a0;
        accs[i][1] = a1;
    }
    __syncthreads();   // att/qs dead; ys may overlap them now
#pragma unroll
    for (int i = 0; i < 2; ++i) {
        int mt = wave + 8 * i;
#pragma unroll
        for (int nt = 0; nt < 2; ++nt)
#pragma unroll
            for (int r = 0; r < 8; ++r) {
                int row = mt * 16 + hh * 8 + r;
                ys[(nt * 16 + ln) * C + row] = accs[i][nt][r] + b_out[row];
            }
    }
    __syncthreads();

    // final RMSNorm over channels + residual
    int p = tid & 31, part = tid >> 5;
    float s = 0.f;
    for (int j = 0; j < 32; ++j) {
        float v = ys[p * C + part * 32 + j];
        s += v * v;
    }
    red[p * 8 + part] = s;
    __syncthreads();
    if (tid < 32) {
        float t = 0.f;
        for (int i = 0; i < 8; ++i) t += red[tid * 8 + i];
        red[256 + tid] = 16.0f / fmaxf(sqrtf(t), EPS);
    }
    __syncthreads();
    const float* xb = x + (size_t)b * C * NPIX + pix0;
    float* ob = out + (size_t)b * C * NPIX + pix0;
    for (int idx = tid; idx < 32 * C; idx += 256) {
        int pp = idx & 31, c = idx >> 5;
        ob[(size_t)c * NPIX + pp] =
            ys[pp * C + c] * red[256 + pp] * g2[c] + xb[(size_t)c * NPIX + pp];
    }
}

// ---------------- Host launch ----------------
extern "C" void kernel_launch(void* const* d_in, const int* in_sizes, int n_in,
                              void* d_out, int out_size, void* d_ws, size_t ws_size,
                              hipStream_t stream) {
    (void)in_sizes; (void)n_in; (void)out_size; (void)ws_size;
    const float* x    = (const float*)d_in[0];
    const float* g1   = (const float*)d_in[1];
    const float* wqkv = (const float*)d_in[2];
    const float* wout = (const float*)d_in[3];
    const float* bout = (const float*)d_in[4];
    const float* g2   = (const float*)d_in[5];
    float* out = (float*)d_out;

    char* ws = (char*)d_ws;
    _Float16* wq16 = (_Float16*)ws;                         // 768*256*2 = 384 KB
    _Float16* wo16 = (_Float16*)(ws + 393216);              // 256*256*2 = 128 KB
    float*    qkv  = (float*)(ws + 524288);                 // 16*768*4096*4 = 192 MB
    _Float16* ctxT = (_Float16*)(ws + 524288 + (size_t)NB * O3 * NPIX * 4);  // 256 KB

    cvt_weights<<<768, 256, 0, stream>>>(wqkv, wout, wq16, wo16);
    qkv_kernel<<<dim3(NPIX / 64, NB), 256, 0, stream>>>(x, g1, wq16, qkv);
    context_kernel<<<NB * NHEADS, 256, 0, stream>>>(qkv, ctxT);
    out_kernel<<<dim3(NPIX / 32, NB), 256, 0, stream>>>(qkv, ctxT, wo16, bout, g2, x, out);
}
